// Model_27676769255432
// MI455X (gfx1250) — matrix operand order repaired and checked by reading, not compiled, not run on hardware
//
#include <hip/hip_runtime.h>
#include <hip/hip_bf16.h>
#include <math.h>

// MI455X (gfx1250) implementation.
// ~50 GFLOP of GEMM over ~60MB of params -> compute-bound; all GEMMs go
// through v_wmma_f32_16x16x32_bf16 (bf16 in, fp32 accumulate), 2x2 register
// blocking (32x32 C-tile per wave). Conv3d layers (1x1 spatial, K<=224) stay
// on fp32 VALU (negligible FLOPs). f2 row staging uses the Tensor Data Mover.

typedef __bf16 bf16;
typedef __attribute__((ext_vector_type(16))) __bf16 v16bf;
typedef __attribute__((ext_vector_type(8)))  __bf16 v8bf;
typedef __attribute__((ext_vector_type(8)))  float  v8f;
typedef __attribute__((ext_vector_type(4)))  unsigned int v4u;
typedef __attribute__((ext_vector_type(8)))  int v8i;
typedef __attribute__((ext_vector_type(4)))  int v4i;

// Load a 16-bit A-matrix WMMA fragment: 16 contiguous bf16 (32B) at p.
static __device__ __forceinline__ v16bf load_frag_a(const bf16* p)
{
  v8bf lo = *(const v8bf*)(p);
  v8bf hi = *(const v8bf*)(p + 8);
  v16bf r;
#pragma unroll
  for (int i = 0; i < 8; ++i) { r[i] = lo[i]; r[i + 8] = hi[i]; }
  return r;
}

// Load a pre-swizzled B fragment: 16 contiguous bf16 (32B) per lane.
static __device__ __forceinline__ v16bf load_frag_b(const bf16* p)
{
  v8bf lo = *(const v8bf*)(p);
  v8bf hi = *(const v8bf*)(p + 8);
  v16bf r;
#pragma unroll
  for (int i = 0; i < 8; ++i) { r[i] = lo[i]; r[i + 8] = hi[i]; }
  return r;
}

static __device__ __forceinline__ v8f wmma_bf16(v16bf a, v16bf b, v8f c)
{
  return __builtin_amdgcn_wmma_f32_16x16x32_bf16(
      false, a, false, b, (short)0, c, false, false);
}

// ---------------------------------------------------------------------------
// WMMA GEMM: C[M,N] = A[M,K] * B[K,N] (+bias[N], optional relu)
//   A: bf16 row-major. K % 32 == 0, M % 128 == 0, N % 32 == 0.
//   Bsw: bf16 pre-swizzled per-lane fragment order:
//     Bsw[(((kt*(N/16))+nt)*32 + lane)*16 + e]
//       = B[kt*32 + (lane>>4)*16 + e][nt*16 + (lane&15)]
//   One wave computes a 32x32 C tile (2x2 WMMA blocking); block = 4 waves.
// ---------------------------------------------------------------------------
__global__ __launch_bounds__(128) void gemm_bf16_wmma(
    const bf16* __restrict__ A, const bf16* __restrict__ Bsw,
    const float* __restrict__ bias, float* __restrict__ C,
    int M, int N, int K, int relu)
{
  (void)M;
  const int lane = threadIdx.x & 31;
  const int tm   = (blockIdx.y * 4 + threadIdx.y) * 32;
  const int tn   = blockIdx.x * 32;
  const int half = lane >> 4;
  const int lm   = lane & 15;
  const int ntiles = N >> 4;

  const bf16* a0p = A + (size_t)(tm + lm) * K + half * 16;
  const bf16* a1p = a0p + (size_t)16 * K;
  const bf16* b0p = Bsw + ((size_t)blockIdx.x * 2 * 32 + lane) * 16;
  const bf16* b1p = b0p + 512;                  // next n-tile, same k-tile
  const size_t bstep = (size_t)ntiles * 512;    // elems per k-tile step

  v8f acc00 = {}, acc01 = {}, acc10 = {}, acc11 = {};
  for (int k0 = 0; k0 < K; k0 += 32) {
    v16bf a0 = load_frag_a(a0p);
    v16bf a1 = load_frag_a(a1p);
    v16bf b0 = load_frag_b(b0p);
    v16bf b1 = load_frag_b(b1p);
    acc00 = wmma_bf16(a0, b0, acc00);
    acc01 = wmma_bf16(a0, b1, acc01);
    acc10 = wmma_bf16(a1, b0, acc10);
    acc11 = wmma_bf16(a1, b1, acc11);
    a0p += 32; a1p += 32;
    b0p += bstep; b1p += bstep;
  }

  // C/D f32 layout: element i -> row (i + 8*half), col = lane&15
  const int n0 = tn + lm, n1 = tn + 16 + lm;
  const float bv0 = bias ? bias[n0] : 0.0f;
  const float bv1 = bias ? bias[n1] : 0.0f;
#pragma unroll
  for (int i = 0; i < 8; ++i) {
    const int m0 = tm + half * 8 + i;
    const int m1 = m0 + 16;
    float v00 = acc00[i] + bv0, v01 = acc01[i] + bv1;
    float v10 = acc10[i] + bv0, v11 = acc11[i] + bv1;
    if (relu) {
      v00 = fmaxf(v00, 0.0f); v01 = fmaxf(v01, 0.0f);
      v10 = fmaxf(v10, 0.0f); v11 = fmaxf(v11, 0.0f);
    }
    C[(size_t)m0 * N + n0] = v00;
    C[(size_t)m0 * N + n1] = v01;
    C[(size_t)m1 * N + n0] = v10;
    C[(size_t)m1 * N + n1] = v11;
  }
}

// Pack W[N][K] (fp32, row-major "weight[out][in]") into swizzled bf16 B^T.
__global__ void pack_b_wT(const float* __restrict__ W, bf16* __restrict__ Bsw,
                          int K, int N)
{
  size_t idx = (size_t)blockIdx.x * blockDim.x + threadIdx.x;
  size_t total = (size_t)K * N;
  if (idx >= total) return;
  int e = (int)(idx & 15);
  int l = (int)((idx >> 4) & 31);
  size_t tile = idx >> 9;
  int ntiles = N >> 4;
  int nt = (int)(tile % ntiles);
  int kt = (int)(tile / ntiles);
  int k = kt * 32 + (l >> 4) * 16 + e;
  int n = nt * 16 + (l & 15);
  Bsw[idx] = (bf16)W[(size_t)n * K + k];
}

// ---------------------------------------------------------------------------
// Conv3d (kernel kd x 1 x 1, VALID, stride strd in depth) + bias + relu.
// X: [256][Cin][Din][25], W: [Cout][Cin][kd], Y: [256][Cout][Dout][25]
// ---------------------------------------------------------------------------
__global__ void conv3d_relu(const float* __restrict__ X, const float* __restrict__ W,
                            const float* __restrict__ B, float* __restrict__ Y,
                            int Cin, int Cout, int Din, int Dout, int kd, int strd)
{
  size_t idx = (size_t)blockIdx.x * blockDim.x + threadIdx.x;
  size_t total = (size_t)256 * Cout * Dout * 25;
  if (idx >= total) return;
  int s  = (int)(idx % 25);
  int d  = (int)((idx / 25) % Dout);
  int co = (int)((idx / ((size_t)25 * Dout)) % Cout);
  int n  = (int)(idx / ((size_t)25 * Dout * Cout));
  float acc = B[co];
  const int d0 = d * strd;
  const float* xn = X + (size_t)n * Cin * Din * 25 + s;
  const float* wc = W + (size_t)co * Cin * kd;
  for (int ci = 0; ci < Cin; ++ci) {
#pragma unroll 4
    for (int t = 0; t < kd; ++t)
      acc += xn[((size_t)ci * Din + d0 + t) * 25] * wc[ci * kd + t];
  }
  Y[idx] = fmaxf(acc, 0.0f);
}

// im2col for 2x2 VALID conv over [256][64*Dd][5][5] -> A[4096][64*Dd*4] bf16
__global__ void im2col_c2(const float* __restrict__ X, bf16* __restrict__ A, int Dd)
{
  int K = 64 * Dd * 4;
  size_t idx = (size_t)blockIdx.x * blockDim.x + threadIdx.x;
  size_t total = (size_t)4096 * K;
  if (idx >= total) return;
  int k = (int)(idx % K);
  int m = (int)(idx / K);
  int n = m >> 4, oh = (m >> 2) & 3, ow = m & 3;
  int kw = k & 1, kh = (k >> 1) & 1;
  int c3 = k >> 2;
  int d = c3 % Dd, c = c3 / Dd;
  A[idx] = (bf16)X[(((size_t)n * 64 + c) * Dd + d) * 25 + (oh + kh) * 5 + (ow + kw)];
}

// Conv2d GEMM output C[(n,oh,ow)][o] -> flattened NCHW A[n][o*16 + oh*4 + ow]
__global__ void permute_to_flat(const float* __restrict__ Cv, bf16* __restrict__ A, int O)
{
  size_t idx = (size_t)blockIdx.x * blockDim.x + threadIdx.x;
  size_t total = (size_t)256 * O * 16;
  if (idx >= total) return;
  int r = (int)(idx % ((size_t)O * 16));
  int n = (int)(idx / ((size_t)O * 16));
  int o = r >> 4, s = r & 15;
  A[idx] = (bf16)Cv[((size_t)n * 16 + s) * O + o];
}

__global__ void cvt_bf16(const float* __restrict__ X, bf16* __restrict__ Y, size_t n)
{
  size_t i = (size_t)blockIdx.x * blockDim.x + threadIdx.x;
  if (i < n) Y[i] = (bf16)X[i];
}

__global__ void elem_mul(const float* __restrict__ A, const float* __restrict__ B,
                         float* __restrict__ C, size_t n)
{
  size_t i = (size_t)blockIdx.x * blockDim.x + threadIdx.x;
  if (i < n) C[i] = A[i] * B[i];
}

__global__ void row_norm2(const float* __restrict__ X, float* __restrict__ Nrm, int cols)
{
  int r = blockIdx.x * blockDim.x + threadIdx.x;
  if (r >= 256) return;
  float acc = 0.0f;
  const float* xr = X + (size_t)r * cols;
  for (int k = 0; k < cols; ++k) acc += xr[k] * xr[k];
  Nrm[r] = acc;
}

// f[i][j] = sqrt(|a_i|^2 + |b_j|^2 - 2 * (a.b^T)[i][j])
__global__ void f_finalize(const float* __restrict__ G, const float* __restrict__ NA,
                           const float* __restrict__ NB, float* __restrict__ F)
{
  int idx = blockIdx.x * blockDim.x + threadIdx.x;  // 65536
  int i = idx >> 8, j = idx & 255;
  F[idx] = sqrtf(NA[i] + NB[j] - 2.0f * G[idx]);
}

// ---------------------------------------------------------------------------
// f2[i][j] = ||t_i - t_j||_2 via explicit difference (faithful to reference).
// Row i (512 floats, 2KB) is DMA'd into LDS by the Tensor Data Mover:
// 1-row 2D tile, data_size=4B, issued by wave 0 only (TDM ignores EXEC, so
// waves 1..7 branch around the issue entirely), completion via TENSORcnt.
// Uses dynamic LDS so the tile's lds_addr is 0.
// ---------------------------------------------------------------------------
__global__ void dist_diff(const float* __restrict__ T, float* __restrict__ F2)
{
  extern __shared__ float ai[];                 // 512 floats, LDS offset 0
  const int i = blockIdx.x;                     // 256 rows
  const int j = threadIdx.x;                    // 256 threads

  if (threadIdx.x < 32) {                       // wave 0 issues the TDM op
    unsigned long long ga = (unsigned long long)(const void*)(T + (size_t)i * 512);
    v4u g0 = {0u, 0u, 0u, 0u};
    g0[0] = 1u;                                 // count=1, user descriptor
    g0[1] = 0u;                                 // lds_addr = 0 (dynamic LDS)
    g0[2] = (unsigned int)ga;                   // global_addr[31:0]
    g0[3] = (unsigned int)(ga >> 32) | (2u << 30); // global_addr[56:32] | type=2
    v8i g1 = {0, 0, 0, 0, 0, 0, 0, 0};
    g1[0] = (2 << 16);                          // data_size = 4 bytes
    g1[1] = (512 << 16);                        // tensor_dim0[15:0] = 512
    g1[2] = (1 << 16);                          // tensor_dim1[15:0] = 1
    g1[3] = (512 << 16);                        // tile_dim0 = 512
    g1[4] = 1;                                  // tile_dim1 = 1
    g1[5] = 512;                                // tensor_dim0_stride = 512
    v4i gz4 = {0, 0, 0, 0};
    v8i gz8 = {0, 0, 0, 0, 0, 0, 0, 0};
    __builtin_amdgcn_tensor_load_to_lds(g0, g1, gz4, gz4, gz8, 0);
    __builtin_amdgcn_s_wait_tensorcnt(0);
  }
  __syncthreads();

  const float* tj = T + (size_t)j * 512;
  float acc = 0.0f;
  for (int k = 0; k < 512; ++k) { float d = ai[k] - tj[k]; acc += d * d; }
  F2[(size_t)i * 256 + j] = sqrtf(acc);
}

// Two-stream head: log_softmax(relu(mod) @ W^T + b), W:[16][512]
__global__ void cls_head(const float* __restrict__ Mod, const float* __restrict__ W,
                         const float* __restrict__ B, float* __restrict__ Out)
{
  const int r = blockIdx.x;       // 256 rows
  const int lane = threadIdx.x;   // 32 (one wave)
  __shared__ float sl[16];
  if (lane < 16) {
    float acc = B[lane];
    const float* wr = W + (size_t)lane * 512;
    const float* mr = Mod + (size_t)r * 512;
    for (int k = 0; k < 512; ++k) acc += fmaxf(mr[k], 0.0f) * wr[k];
    sl[lane] = acc;
  }
  __syncthreads();
  if (lane < 16) {
    float mx = -3.0e38f;
    for (int j = 0; j < 16; ++j) mx = fmaxf(mx, sl[j]);
    float s = 0.0f;
    for (int j = 0; j < 16; ++j) s += expf(sl[j] - mx);
    Out[(size_t)r * 16 + lane] = sl[lane] - mx - logf(s);
  }
}

// ---------------------------------------------------------------------------
// Host orchestration
// ---------------------------------------------------------------------------
static inline dim3 grid1(size_t total, int blk) {
  return dim3((unsigned)((total + (size_t)blk - 1) / blk));
}

static void launch_gemm(const bf16* A, const bf16* B, const float* bias, float* C,
                        int M, int N, int K, int relu, hipStream_t s)
{
  dim3 blk(32, 4);
  dim3 grd((unsigned)(N / 32), (unsigned)(M / 128));
  gemm_bf16_wmma<<<grd, blk, 0, s>>>(A, B, bias, C, M, N, K, relu);
}

extern "C" void kernel_launch(void* const* d_in, const int* in_sizes, int n_in,
                              void* d_out, int out_size, void* d_ws, size_t ws_size,
                              hipStream_t stream)
{
  (void)in_sizes; (void)n_in; (void)out_size; (void)ws_size;

  const float* x1 = (const float*)d_in[0];   // [256,1,103,5,5]
  const float* x2 = (const float*)d_in[1];   // [256,1,72,5,5]
  const float* Sc3w[4] = {(const float*)d_in[3], (const float*)d_in[5],
                          (const float*)d_in[7], (const float*)d_in[9]};
  const float* Sc3b[4] = {(const float*)d_in[4], (const float*)d_in[6],
                          (const float*)d_in[8], (const float*)d_in[10]};
  const float* Sc2w = (const float*)d_in[11];
  const float* Sc2b = (const float*)d_in[12];
  const float* Sl1w = (const float*)d_in[13];
  const float* Sl1b = (const float*)d_in[14];
  const float* Sl2w = (const float*)d_in[15];
  const float* Sl2b = (const float*)d_in[16];
  const float* Tc3w[4] = {(const float*)d_in[17], (const float*)d_in[19],
                          (const float*)d_in[21], (const float*)d_in[23]};
  const float* Tc3b[4] = {(const float*)d_in[18], (const float*)d_in[20],
                          (const float*)d_in[22], (const float*)d_in[24]};
  const float* Tc2w = (const float*)d_in[25];
  const float* Tc2b = (const float*)d_in[26];
  const float* Tl1w = (const float*)d_in[27];
  const float* Tl1b = (const float*)d_in[28];
  const float* Tl2w = (const float*)d_in[29];
  const float* Tl2b = (const float*)d_in[30];
  const float* att3w = (const float*)d_in[31];
  const float* att3b = (const float*)d_in[32];
  const float* att4w = (const float*)d_in[33];
  const float* att4b = (const float*)d_in[34];
  const float* tsw   = (const float*)d_in[35];
  const float* tsb   = (const float*)d_in[36];

  // Workspace layout (bump allocator, 256B aligned)
  char* ws = (char*)d_ws;
  size_t off = 0;
  auto bump = [&](size_t bytes) -> char* {
    char* p = ws + off;
    off += (bytes + 255) & ~(size_t)255;
    return p;
  };
  float* P0 = (float*)bump((size_t)10649600 * 4);  // max conv3d stage (T4)
  float* P1 = (float*)bump((size_t)10649600 * 4);
  bf16*  AB = (bf16*) bump((size_t)27262976 * 2);  // max A (T im2col 4096x6656)
  bf16*  BB = (bf16*) bump((size_t)16777216 * 2);  // max B (T_l1 8192x2048)
  float* CB = (float*)bump((size_t)2097152 * 4);   // max C (4096x512)
  float* SRC_OUT = (float*)bump((size_t)131072 * 4);
  float* TGT_OUT = (float*)bump((size_t)131072 * 4);
  float* SRC_MOD = (float*)bump((size_t)131072 * 4);
  float* TGT_MOD = (float*)bump((size_t)131072 * 4);
  float* NAb = (float*)bump(256 * 4);
  float* NBb = (float*)bump(256 * 4);

  // Output layout (flat, return order)
  float* out_scls = (float*)d_out;           // [256,16]
  float* out_tcls = out_scls + 4096;         // [256,16]
  float* out_satt = out_tcls + 4096;         // [256,512]
  float* out_tatt = out_satt + 131072;       // [256,512]
  float* out_f    = out_tatt + 131072;       // [256,256]
  float* out_f2   = out_f + 65536;           // [256,256]

  // One HybridSN stream: conv3d x4 -> im2col -> conv2d GEMM -> l1 -> l2
  auto run_stream = [&](const float* x, int Din0, int kd, const int* strides,
                        const float* const* c3w, const float* const* c3b,
                        const float* c2w, const float* c2b, int Ocv,
                        const float* l1w, const float* l1b,
                        const float* l2w, const float* l2b, float* OUT)
  {
    int Cin = 1, Din = Din0, Dd = 0;
    const float* src = x;
    float* dst = P0;
    for (int i = 0; i < 4; ++i) {
      int Cout = 8 << i;
      int Dout = (Din - kd) / strides[i] + 1;
      size_t total = (size_t)256 * Cout * Dout * 25;
      conv3d_relu<<<grid1(total, 256), 256, 0, stream>>>(
          src, c3w[i], c3b[i], dst, Cin, Cout, Din, Dout, kd, strides[i]);
      src = dst;
      dst = (dst == P0) ? P1 : P0;
      Cin = Cout; Din = Dout; Dd = Dout;
    }
    // src == P1: [256][64][Dd][25]
    const int K1 = 64 * Dd * 4;
    im2col_c2<<<grid1((size_t)4096 * K1, 256), 256, 0, stream>>>(src, AB, Dd);
    pack_b_wT<<<grid1((size_t)K1 * Ocv, 256), 256, 0, stream>>>(c2w, BB, K1, Ocv);
    launch_gemm(AB, BB, c2b, CB, 4096, Ocv, K1, /*relu=*/1, stream);

    permute_to_flat<<<grid1((size_t)256 * Ocv * 16, 256), 256, 0, stream>>>(CB, AB, Ocv);
    const int K2 = Ocv * 16;                          // 4096 (S) / 8192 (T)
    pack_b_wT<<<grid1((size_t)K2 * 2048, 256), 256, 0, stream>>>(l1w, BB, K2, 2048);
    launch_gemm(AB, BB, l1b, CB, 256, 2048, K2, /*relu=*/1, stream);

    cvt_bf16<<<grid1((size_t)256 * 2048, 256), 256, 0, stream>>>(CB, AB, (size_t)256 * 2048);
    pack_b_wT<<<grid1((size_t)2048 * 512, 256), 256, 0, stream>>>(l2w, BB, 2048, 512);
    launch_gemm(AB, BB, l2b, OUT, 256, 512, 2048, /*relu=*/0, stream);
  };

  // localattention: relu(x@l3^T+b) @ l4^T + b
  auto run_att = [&](const float* OUT, float* ATT)
  {
    cvt_bf16<<<grid1(131072, 256), 256, 0, stream>>>(OUT, AB, 131072);
    pack_b_wT<<<grid1((size_t)512 * 256, 256), 256, 0, stream>>>(att3w, BB, 512, 256);
    launch_gemm(AB, BB, att3b, CB, 256, 256, 512, /*relu=*/1, stream);
    cvt_bf16<<<grid1(65536, 256), 256, 0, stream>>>(CB, AB, 65536);
    pack_b_wT<<<grid1((size_t)256 * 512, 256), 256, 0, stream>>>(att4w, BB, 256, 512);
    launch_gemm(AB, BB, att4b, ATT, 256, 512, 256, /*relu=*/0, stream);
  };

  static const int s_str[4] = {2, 1, 2, 1};
  static const int t_str[4] = {2, 1, 1, 1};

  // Source stream: 103 -> 49 -> 43 -> 19 -> 13; conv2d K=3328, O=256
  run_stream(x1, 103, 7, s_str, Sc3w, Sc3b, Sc2w, Sc2b, 256,
             Sl1w, Sl1b, Sl2w, Sl2b, SRC_OUT);
  // Target stream: 72 -> 35 -> 32 -> 29 -> 26; conv2d K=6656, O=512
  run_stream(x2, 72, 4, t_str, Tc3w, Tc3b, Tc2w, Tc2b, 512,
             Tl1w, Tl1b, Tl2w, Tl2b, TGT_OUT);

  run_att(SRC_OUT, out_satt);
  run_att(TGT_OUT, out_tatt);

  // Modulated features
  elem_mul<<<grid1(131072, 256), 256, 0, stream>>>(SRC_OUT, out_satt, SRC_MOD, 131072);
  elem_mul<<<grid1(131072, 256), 256, 0, stream>>>(TGT_OUT, out_tatt, TGT_MOD, 131072);

  // f = EuclideanDistances(src_mod, tgt_mod): gram via WMMA, then finalize
  row_norm2<<<1, 256, 0, stream>>>(SRC_MOD, NAb, 512);
  row_norm2<<<1, 256, 0, stream>>>(TGT_MOD, NBb, 512);
  cvt_bf16<<<grid1(131072, 256), 256, 0, stream>>>(SRC_MOD, AB, 131072);
  pack_b_wT<<<grid1((size_t)512 * 256, 256), 256, 0, stream>>>(TGT_MOD, BB, 512, 256);
  launch_gemm(AB, BB, nullptr, CB, 256, 256, 512, /*relu=*/0, stream);
  f_finalize<<<256, 256, 0, stream>>>(CB, NAb, NBb, out_f);

  // f2 = Distances(tgt_mod, tgt_mod) with TDM row staging (2KB dynamic LDS)
  dist_diff<<<256, 256, 2048, stream>>>(TGT_MOD, out_f2);

  // Two-stream log-softmax heads
  cls_head<<<256, 32, 0, stream>>>(SRC_MOD, tsw, tsb, out_scls);
  cls_head<<<256, 32, 0, stream>>>(TGT_MOD, tsw, tsb, out_tcls);
}
